// ResidualVectorQuantizer_20280835572427
// MI455X (gfx1250) — matrix operand order repaired and checked by reading, not compiled, not run on hardware
//
#include <hip/hip_runtime.h>
#include <hip/hip_bf16.h>

// ---------------------------------------------------------------------------
// Residual Vector Quantizer for MI455X (gfx1250, wave32, WMMA).
//
// Shapes (fixed by reference): x[8,1024,512] f32, codebooks[3,1024,512] f32.
// N=8192 tokens, D=512, K=1024 codes, LEVELS=3.
//
// Distance GEMM (25.8 GFLOP total) runs on v_wmma_f32_16x16x32_f16.
// |r|^2 and |c|^2 terms are exact f32; only the cross term r.c is f16.
// Phase B is software-pipelined: fragments for k-step ks+1 are fetched into a
// second register set while the two WMMAs of k-step ks execute (two col-tiles
// per wave share each A fragment -> 2 wmma per ds-load pair, 2 indep chains).
// ---------------------------------------------------------------------------

typedef _Float16 half_t;
typedef __attribute__((ext_vector_type(16))) _Float16 v16h;
typedef __attribute__((ext_vector_type(8)))  _Float16 v8h;
typedef __attribute__((ext_vector_type(8)))  float    v8f;

union U16H { v16h v; v8h h[2]; };

#define N_TOK   8192
#define DIM     512
#define KCB     1024
#define LEVELS  3
#define ROWS    16          // tokens per workgroup
#define THREADS 256         // 8 waves of 32

// LDS: aTile f16[16*512] + d2 f32[16*1024] + stats + reduce scratch
#define SMEM_BYTES (ROWS*DIM*2 + ROWS*KCB*4 + 4*ROWS*4 + THREADS*4)

// A frag (16-bit A 16x32): lane m = l%16; K {0..7,16..23} / {8..15,24..31}
__device__ __forceinline__ U16H loadA(const half_t* aTile, int mloc, int hi, int k0) {
  const half_t* ap = aTile + mloc * DIM + k0 + hi * 8;
  U16H a;
  a.h[0] = *reinterpret_cast<const v8h*>(ap);
  a.h[1] = *reinterpret_cast<const v8h*>(ap + 16);
  return a;
}

// B frag (16-bit B 32x16): lane n = l%16; K {0..7,16..23} / {8..15,24..31}
__device__ __forceinline__ U16H loadB(const half_t* bBase, int hi, int k0) {
  const half_t* bp = bBase + k0 + hi * 8;
  U16H b;
  b.h[0] = *reinterpret_cast<const v8h*>(bp);
  b.h[1] = *reinterpret_cast<const v8h*>(bp + 16);
  return b;
}

// ---------------------------------------------------------------------------
// Kernel 0: codebook f32 -> f16, and c2[k] = sum(c^2) in f32.
// ---------------------------------------------------------------------------
__global__ void rvq_prep(const float* __restrict__ cbf,
                         half_t* __restrict__ cbh,
                         float* __restrict__ c2) {
  __shared__ float red[128];
  const int row = blockIdx.x;              // 0 .. 3071
  const int t   = threadIdx.x;             // 0 .. 127
  const float* src = cbf + (size_t)row * DIM;
  half_t*      dst = cbh + (size_t)row * DIM;
  float ss = 0.0f;
#pragma unroll
  for (int i = 0; i < 4; ++i) {
    const int j = t * 4 + i;
    const float v = src[j];
    ss += v * v;
    dst[j] = (half_t)v;
  }
  red[t] = ss;
  __syncthreads();
  for (int s = 64; s > 0; s >>= 1) {
    if (t < s) red[t] += red[t + s];
    __syncthreads();
  }
  if (t == 0) c2[row] = red[0];
}

// ---------------------------------------------------------------------------
// Kernel 1: zero per-call accumulators (ws is NOT re-poisoned between replays).
// ---------------------------------------------------------------------------
__global__ void rvq_zero(float* __restrict__ probsum,
                         float* __restrict__ commitacc) {
  const int i = blockIdx.x * blockDim.x + threadIdx.x;
  if (i < LEVELS * KCB) probsum[i] = 0.0f;
  if (i == 0) commitacc[0] = 0.0f;
}

// ---------------------------------------------------------------------------
// Kernel 2: one RVQ level for a 16-token tile.
// ---------------------------------------------------------------------------
__global__ void __launch_bounds__(THREADS)
rvq_level(const int level,
          const float* __restrict__ resIn,     // x (level 0) or res buffer
          float* __restrict__ resOut,          // res buffer
          float* __restrict__ qsum,            // running quantized sum
          float* __restrict__ qout,            // d_out quantized_final (last level)
          const half_t* __restrict__ cbh,      // f16 codebooks
          const float* __restrict__ cbf,       // f32 codebooks (exact gather)
          const float* __restrict__ c2g,       // per-code squared norms
          float* __restrict__ probsum,         // [LEVELS*KCB] prob column sums
          float* __restrict__ commitacc,       // scalar sum of diff^2
          float* __restrict__ codesOut,        // d_out codes region (float-valued)
          const int firstLevel,
          const int lastLevel) {
  extern __shared__ char smem[];
  half_t* aTile = (half_t*)smem;                              // 16 x 512 f16
  float*  d2t   = (float*)(smem + ROWS * DIM * 2);            // 16 x 1024 f32
  float*  x2s   = (float*)(smem + ROWS * DIM * 2 + ROWS * KCB * 4);
  float*  mrow  = x2s + ROWS;
  float*  srow  = mrow + ROWS;
  int*    irow  = (int*)(srow + ROWS);
  float*  red   = (float*)(irow + ROWS);                      // [256]

  const int t    = threadIdx.x;
  const int wave = t >> 5;
  const int lane = t & 31;
  const int row0 = blockIdx.x * ROWS;

  // ---------------- Phase A: residual tile -> LDS f16, exact |r|^2 ---------
  {
    const int rowA = t >> 4;          // 0..15
    const int segA = t & 15;          // 0..15, 32 elements each
    const float* rp = resIn + (size_t)(row0 + rowA) * DIM + segA * 32;
    half_t* ap = aTile + rowA * DIM + segA * 32;
    float ss = 0.0f;
#pragma unroll
    for (int i = 0; i < 32; ++i) {
      const float v = rp[i];
      ss += v * v;
      ap[i] = (half_t)v;
    }
    red[t] = ss;
    __syncthreads();
    if (segA == 0) {
      float s = 0.0f;
#pragma unroll
      for (int j = 0; j < 16; ++j) s += red[rowA * 16 + j];
      x2s[rowA] = s;
    }
    __syncthreads();
  }

  // ---------------- Phase B: WMMA distance GEMM (pipelined, 2 col-tiles) ---
  // wave w covers code columns [w*128, w*128+128): 4 col-tile PAIRS of 16+16.
  {
    const int mloc = lane & 15;
    const int hi   = lane >> 4;       // half-wave select per ISA layouts
    for (int cp = 0; cp < 4; ++cp) {
      const int col0 = wave * 128 + cp * 32;
      const int n0   = col0 + mloc;                 // code rows this lane feeds
      const float c2v0 = c2g[level * KCB + n0];
      const float c2v1 = c2g[level * KCB + n0 + 16];
      const half_t* bBase0 = cbh + ((size_t)level * KCB + n0) * DIM;
      const half_t* bBase1 = bBase0 + 16 * DIM;
      if (cp < 3) {  // pull next pair's code rows toward the WGP
        __builtin_prefetch(bBase0 + 32 * DIM, 0, 3);   // -> global_prefetch_b8
        __builtin_prefetch(bBase1 + 32 * DIM, 0, 3);
      }

      v8f acc0 = {}, acc1 = {};
      U16H a  = loadA(aTile, mloc, hi, 0);
      U16H b0 = loadB(bBase0, hi, 0);
      U16H b1 = loadB(bBase1, hi, 0);
#pragma unroll
      for (int ks = 0; ks < 16; ++ks) {
        U16H an = a, bn0 = b0, bn1 = b1;   // dead copies removed when ks==15
        if (ks < 15) {
          const int k1 = (ks + 1) * 32;    // fetch next k-step while computing
          an  = loadA(aTile, mloc, hi, k1);
          bn0 = loadB(bBase0, hi, k1);
          bn1 = loadB(bBase1, hi, k1);
        }
        acc0 = __builtin_amdgcn_wmma_f32_16x16x32_f16(
            false, a.v, false, b0.v, (short)0, acc0, false, false);
        acc1 = __builtin_amdgcn_wmma_f32_16x16x32_f16(
            false, a.v, false, b1.v, (short)0, acc1, false, false);
        a = an; b0 = bn0; b1 = bn1;
      }
      // C/D layout: VGPR r holds M=r (lanes<16) / M=r+8 (lanes>=16), N=l%16
#pragma unroll
      for (int r = 0; r < 8; ++r) {
        const int m = r + hi * 8;
        d2t[m * KCB + n0]      = x2s[m] + c2v0 - 2.0f * acc0[r];
        d2t[m * KCB + n0 + 16] = x2s[m] + c2v1 - 2.0f * acc1[r];
      }
    }
    __syncthreads();
  }

  // ---------------- Phase C: per-row argmin + sum(exp) ---------------------
  {
    const int row = 2 * wave + (lane >> 4);   // 8 waves x 2 rows = 16 rows
    const int lir = lane & 15;
    float bestV = 3.4e38f;
    int   bestI = KCB;
    for (int j = lir; j < KCB; j += 16) {
      const float v = d2t[row * KCB + j];
      if (v < bestV || (v == bestV && j < bestI)) { bestV = v; bestI = j; }
    }
#pragma unroll
    for (int m2 = 8; m2 >= 1; m2 >>= 1) {
      const float ov = __shfl_xor(bestV, m2, 32);
      const int   oi = __shfl_xor(bestI, m2, 32);
      if (ov < bestV || (ov == bestV && oi < bestI)) { bestV = ov; bestI = oi; }
    }
    // softmax: logits=-d2, max logit = -min d2 -> p ~ exp(min_d2 - d2)
    float se = 0.0f;
    for (int j = lir; j < KCB; j += 16)
      se += __expf(bestV - d2t[row * KCB + j]);
#pragma unroll
    for (int m2 = 8; m2 >= 1; m2 >>= 1) se += __shfl_xor(se, m2, 32);
    if (lir == 0) {
      mrow[row] = bestV;
      srow[row] = se;
      irow[row] = bestI;
      codesOut[(size_t)(row0 + row) * LEVELS + level] = (float)bestI;
    }
    __syncthreads();
  }

  // ---------------- Phase D: usage stats (column prob sums) ----------------
  {
    float inv[ROWS];
#pragma unroll
    for (int r = 0; r < ROWS; ++r) inv[r] = 1.0f / srow[r];
    for (int c = t; c < KCB; c += THREADS) {
      float acc = 0.0f;
#pragma unroll
      for (int r = 0; r < ROWS; ++r)
        acc += __expf(mrow[r] - d2t[r * KCB + c]) * inv[r];
      atomicAdd(&probsum[level * KCB + c], acc);
    }
  }

  // ---------------- Phase E: gather q, update residual/qsum, commit --------
  {
    const int rowE = t >> 4;
    const int off  = (t & 15) * 32;
    const int qi   = irow[rowE];
    const float* qp = cbf + ((size_t)level * KCB + qi) * DIM + off;
    const size_t base = (size_t)(row0 + rowE) * DIM + off;
    const float* rp = resIn + base;
    float* ro = resOut + base;
    float* qs = qsum + base;
    float cacc = 0.0f;
#pragma unroll
    for (int i = 0; i < 32; ++i) {
      const float q  = qp[i];
      const float nv = rp[i] - q;        // diff = residual - q (new residual)
      cacc += nv * nv;
      ro[i] = nv;
      const float qv = firstLevel ? q : (qs[i] + q);
      qs[i] = qv;
      if (lastLevel) qout[base + i] = qv;  // quantized_final == qsum (STE)
    }
    red[t] = cacc;
    __syncthreads();
    for (int s = 128; s > 0; s >>= 1) {
      if (t < s) red[t] += red[t + s];
      __syncthreads();
    }
    if (t == 0) atomicAdd(commitacc, red[0]);
  }
}

// ---------------------------------------------------------------------------
// Kernel 3: losses.
// ---------------------------------------------------------------------------
__global__ void rvq_finalize(const float* __restrict__ probsum,
                             const float* __restrict__ commitacc,
                             float* __restrict__ lossOut) {
  __shared__ float fr[THREADS];
  const int t = threadIdx.x;
  float usage = 0.0f;
  for (int lv = 0; lv < LEVELS; ++lv) {
    float local = 0.0f;
    for (int k = t; k < KCB; k += THREADS) {
      float p = probsum[lv * KCB + k] * (1.0f / (float)N_TOK);
      p = fmaxf(p, 1e-5f);
      local += p * __logf(p);
    }
    fr[t] = local;
    __syncthreads();
    for (int s = 128; s > 0; s >>= 1) {
      if (t < s) fr[t] += fr[t + s];
      __syncthreads();
    }
    if (t == 0) {
      const float H = -fr[0];
      usage += 1e-3f * (6.9314718055994531f /* ln(1024) */ - H);
    }
    __syncthreads();
  }
  if (t == 0) {
    lossOut[0] = 0.25f * commitacc[0] * (1.0f / ((float)N_TOK * DIM)) / (float)LEVELS;
    lossOut[1] = usage / (float)LEVELS;
  }
}

// ---------------------------------------------------------------------------
// Launcher. ws layout (~36.8 MB): res | qsum | cbh(f16) | c2 | probsum | commit
// ---------------------------------------------------------------------------
extern "C" void kernel_launch(void* const* d_in, const int* in_sizes, int n_in,
                              void* d_out, int out_size, void* d_ws, size_t ws_size,
                              hipStream_t stream) {
  (void)in_sizes; (void)n_in; (void)out_size; (void)ws_size;
  const float* x   = (const float*)d_in[0];   // [8,1024,512]
  const float* cbf = (const float*)d_in[1];   // [3,1024,512]

  char* ws = (char*)d_ws;
  const size_t ND = (size_t)N_TOK * DIM;
  float*  res       = (float*)(ws);
  float*  qsum      = (float*)(ws + ND * 4);
  half_t* cbh       = (half_t*)(ws + 2 * ND * 4);
  float*  c2        = (float*)(ws + 2 * ND * 4 + (size_t)LEVELS * KCB * DIM * 2);
  float*  probsum   = c2 + LEVELS * KCB;
  float*  commitacc = probsum + LEVELS * KCB;

  float* qout   = (float*)d_out;              // [N,D]
  float* codes  = qout + ND;                  // [N,LEVELS] as float-valued ints
  float* losses = codes + (size_t)N_TOK * LEVELS;  // commit, usage

  rvq_prep<<<LEVELS * KCB, 128, 0, stream>>>(cbf, cbh, c2);
  rvq_zero<<<(LEVELS * KCB + 255) / 256, 256, 0, stream>>>(probsum, commitacc);

  for (int lv = 0; lv < LEVELS; ++lv) {
    const float* rin = (lv == 0) ? x : res;
    rvq_level<<<N_TOK / ROWS, THREADS, SMEM_BYTES, stream>>>(
        lv, rin, res, qsum, qout, cbh, cbf, c2, probsum, commitacc, codes,
        lv == 0 ? 1 : 0, lv == LEVELS - 1 ? 1 : 0);
  }

  rvq_finalize<<<1, THREADS, 0, stream>>>(probsum, commitacc, losses);
}
